// JointNetwork_2044404433374
// MI455X (gfx1250) — matrix operand order repaired and checked by reading, not compiled, not run on hardware
//
#include <hip/hip_runtime.h>

#define D_JOINT 640
#define VOCAB   1024
#define KIN     512
#define B_      8
#define T_      256
#define U_      64
#define GP      72            // LDS row pitch in halfs (64 data + 8 pad) -> conflict-free b128

typedef __attribute__((ext_vector_type(8)))  _Float16 v8h;
typedef __attribute__((ext_vector_type(16))) _Float16 v16h;
typedef __attribute__((ext_vector_type(8)))  float    v8f;
typedef __attribute__((ext_vector_type(4)))  int      v4i;

// ---- CDNA5 async global->LDS copy (ASYNCcnt), with safe fallback ----------
#if defined(__HIP_DEVICE_COMPILE__) && __has_builtin(__builtin_amdgcn_global_load_async_to_lds_b128)
#define USE_ASYNC_LDS 1
#define ASYNC_CP16(gsrc, ldst)                                                \
    __builtin_amdgcn_global_load_async_to_lds_b128(                           \
        (__attribute__((address_space(1))) v4i*)(gsrc),                       \
        (__attribute__((address_space(3))) v4i*)(ldst), 0, 0)
#if __has_builtin(__builtin_amdgcn_s_wait_asynccnt)
#define WAIT_ASYNC(n) __builtin_amdgcn_s_wait_asynccnt(n)
#else
#define WAIT_ASYNC(n) asm volatile("s_wait_asynccnt %0" ::"i"(n))
#endif
#else
#define USE_ASYNC_LDS 0
#define WAIT_ASYNC(n)
#endif

__device__ __forceinline__ float fast_tanh(float x) {
    // tanh(x) = 1 - 2/(exp(2x)+1); v_exp_f32 + v_rcp_f32; saturates at +-inf.
    float e = __builtin_amdgcn_exp2f(x * 2.8853900817779268f); // 2*log2(e)
    return 1.0f - 2.0f * __builtin_amdgcn_rcpf(e + 1.0f);
}

// A-operand fragment (16x32 f16, ISA layout): lane holds row (lane&15);
// halfs[0..7] = K kof..kof+7, halfs[8..15] = K kof+16..kof+23, kof = 8*(lane>=16).
__device__ __forceinline__ v16h ld_frag_a(const _Float16* p) {
    v8h lo = *(const v8h*)p;
    v8h hi = *(const v8h*)(p + 16);
    return __builtin_shufflevector(lo, hi, 0,1,2,3,4,5,6,7,8,9,10,11,12,13,14,15);
}
// B-operand fragment (32x16 f16): lane holds column (lane&15);
// halfs[0..7] = K kb..kb+7, halfs[8..15] = K kb+16..kb+23, kb = 8*(lane>=16).
__device__ __forceinline__ v16h ld_frag_b(const _Float16* p) {
    v8h lo = *(const v8h*)p;
    v8h hi = *(const v8h*)(p + 16);
    return __builtin_shufflevector(lo, hi, 0,1,2,3,4,5,6,7,8,9,10,11,12,13,14,15);
}

__global__ __launch_bounds__(256)
void cvt_f32_to_f16(const float* __restrict__ in, _Float16* __restrict__ out, int n) {
    int i = blockIdx.x * 256 + threadIdx.x;
    if (i < n) out[i] = (_Float16)in[i];
}

// C[M,640] = A_h[M,512] * W_h[640,512]^T   (f32 accum via f16 WMMA)
// grid = (M/64, 640/128), block = 256 (8 waves, 4x2), K-chunks of 64.
__global__ __launch_bounds__(256)
void gemm_fg(const _Float16* __restrict__ A, const _Float16* __restrict__ W,
             float* __restrict__ C) {
    extern __shared__ char smem[];
    _Float16* a_s = (_Float16*)smem;                 // 64 x GP halfs
    _Float16* b_s = (_Float16*)(smem + 64 * GP * 2); // 128 x GP halfs

    const int tid  = threadIdx.x;
    const int lane = tid & 31;
    const int wave = tid >> 5;
    const int wm = wave & 3, wn = wave >> 2;
    const int bx = blockIdx.x, by = blockIdx.y;
    const int l15  = lane & 15;
    const int kofh = (lane >> 4) << 3;  // A: +8 halfs on upper half-wave
    const int kofb = (lane >> 4) << 3;  // B: +8 halfs on upper half-wave

    v8f acc[4] = {};

    for (int kc = 0; kc < KIN; kc += 64) {
        __syncthreads();
        { // A tile: 64 rows x 64 halfs
            int r = tid >> 2, c = (tid & 3) << 4;
            const _Float16* src = A + (size_t)(bx * 64 + r) * KIN + kc + c;
            _Float16* dst = a_s + r * GP + c;
            *(v8h*)dst       = *(const v8h*)src;
            *(v8h*)(dst + 8) = *(const v8h*)(src + 8);
        }
        { // W tile: 128 rows x 64 halfs
            int r = tid >> 1, c = (tid & 1) << 5;
            const _Float16* src = W + (size_t)(by * 128 + r) * KIN + kc + c;
            _Float16* dst = b_s + r * GP + c;
            #pragma unroll
            for (int cc = 0; cc < 32; cc += 8)
                *(v8h*)(dst + cc) = *(const v8h*)(src + cc);
        }
        __syncthreads();
        #pragma unroll
        for (int ks = 0; ks < 64; ks += 32) {
            v16h af = ld_frag_a(a_s + (size_t)((wm << 4) + l15) * GP + ks + kofh);
            #pragma unroll
            for (int j = 0; j < 4; ++j) {
                v16h bf = ld_frag_b(b_s + (size_t)((wn << 6) + (j << 4) + l15) * GP + ks + kofb);
                acc[j] = __builtin_amdgcn_wmma_f32_16x16x32_f16(
                    false, af, false, bf, (short)0, acc[j], false, false);
            }
        }
    }

    const int rlo = (wm << 4) + ((lane >> 4) << 3);
    #pragma unroll
    for (int j = 0; j < 4; ++j) {
        int col = by * 128 + (wn << 6) + (j << 4) + l15;
        #pragma unroll
        for (int v = 0; v < 8; ++v) {
            int row = bx * 64 + rlo + v;
            C[(size_t)row * D_JOINT + col] = acc[j][v];
        }
    }
}

// out[b,t,u,v] = tanh(f[b,t,:] + g[b,u,:]) . W_out[v,:] + b_out[v]
// grid = (B*T, VOCAB/256); block tile 64(U) x 256(V); 8 waves 4x2, each 16x128.
// Double-buffered LDS; next chunk's W_out tile streamed via async global->LDS
// (ASYNCcnt) and next chunk's tanh A-tile built while current chunk's WMMAs run.
__global__ __launch_bounds__(256)
void joint_gemm(const float* __restrict__ f, const float* __restrict__ g,
                const _Float16* __restrict__ Wout, const float* __restrict__ bout,
                float* __restrict__ out) {
    extern __shared__ char smem[];
    float*    f_s = (float*)smem;                               // 640 f32 (2560 B)
    _Float16* a_s = (_Float16*)(smem + 2560);                   // 2 x 64 x GP
    _Float16* b_s = (_Float16*)(smem + 2560 + 2 * 64 * GP * 2); // 2 x 256 x GP

    const int tid  = threadIdx.x;
    const int lane = tid & 31;
    const int wave = tid >> 5;
    const int wm = wave & 3, wn = wave >> 2;
    const int bx = blockIdx.x;            // (b,t) pair: b = bx>>8, t = bx&255
    const int nb = blockIdx.y << 8;       // vocab base (256-wide)
    const int bB = bx >> 8;
    const int l15  = lane & 15;
    const int kofh = (lane >> 4) << 3;
    const int kofb = (lane >> 4) << 3;

    const int ar = tid >> 2;              // u-row this thread stages (0..63)
    const int ac = (tid & 3) << 4;        // k sub-offset (0..48)
    const float*    grow = g + (size_t)(bB * U_ + ar) * D_JOINT;
    const _Float16* wrow = Wout + (size_t)(nb + tid) * D_JOINT; // this thread's vocab row

    // Stage next-chunk W_out tile (256 rows x 64 halfs) into LDS buffer.
    auto fill_b = [&](int kc, _Float16* bbuf) {
        const _Float16* src = wrow + kc;
        _Float16* dst = bbuf + tid * GP;
#if USE_ASYNC_LDS
        #pragma unroll
        for (int cc = 0; cc < 64; cc += 8) ASYNC_CP16(src + cc, dst + cc);
#else
        #pragma unroll
        for (int cc = 0; cc < 64; cc += 8)
            *(v8h*)(dst + cc) = *(const v8h*)(src + cc);
#endif
    };
    // Stage next-chunk A tile: tanh(f + g) for K-range [kc, kc+64), 64x64 f16.
    auto fill_a = [&](int kc, _Float16* abuf) {
        const float* gs = grow + kc + ac;
        const float* fs = f_s + kc + ac;
        v8h h0, h1;
        #pragma unroll
        for (int i = 0; i < 8; ++i) h0[i] = (_Float16)fast_tanh(fs[i] + gs[i]);
        #pragma unroll
        for (int i = 0; i < 8; ++i) h1[i] = (_Float16)fast_tanh(fs[i + 8] + gs[i + 8]);
        _Float16* dst = abuf + ar * GP + ac;
        *(v8h*)dst       = h0;
        *(v8h*)(dst + 8) = h1;
    };

    for (int i = tid; i < D_JOINT; i += 256) f_s[i] = f[(size_t)bx * D_JOINT + i];
    __syncthreads();                       // f_s visible to all threads

    v8f acc[8] = {};

    // Prologue: stage chunk 0 into buffer 0 (8 async issues outstanding/wave).
    fill_b(0, b_s);
    fill_a(0, a_s);

    const int NCHUNK = D_JOINT / 64;       // 10
    for (int ci = 0; ci < NCHUNK; ++ci) {
        const int cur = ci & 1;
        _Float16* a_cur = a_s + cur * (64 * GP);
        _Float16* b_cur = b_s + cur * (256 * GP);
        if (ci + 1 < NCHUNK) {
            // Issue next chunk into the alternate buffers; overlaps this
            // chunk's WMMAs (async B loads in flight through the compute).
            fill_b((ci + 1) * 64, b_s + (1 - cur) * (256 * GP));
            fill_a((ci + 1) * 64, a_s + (1 - cur) * (64 * GP));
            WAIT_ASYNC(8);                 // current chunk's 8 issues complete
        } else {
            WAIT_ASYNC(0);
        }
        __syncthreads();                   // all waves' tiles for `cur` ready

        #pragma unroll
        for (int ks = 0; ks < 64; ks += 32) {
            v16h af = ld_frag_a(a_cur + (size_t)((wm << 4) + l15) * GP + ks + kofh);
            #pragma unroll
            for (int j = 0; j < 8; ++j) {
                v16h bf = ld_frag_b(b_cur + (size_t)((wn << 7) + (j << 4) + l15) * GP + ks + kofb);
                acc[j] = __builtin_amdgcn_wmma_f32_16x16x32_f16(
                    false, af, false, bf, (short)0, acc[j], false, false);
            }
        }
        __syncthreads();                   // protect buffers before next refill
    }

    const size_t Rbase = (size_t)bx * U_;
    const int rlo = (wm << 4) + ((lane >> 4) << 3);
    #pragma unroll
    for (int j = 0; j < 8; ++j) {
        int col = nb + (wn << 7) + (j << 4) + l15;
        float bias = bout[col];
        #pragma unroll
        for (int v = 0; v < 8; ++v) {
            size_t row = Rbase + rlo + v;
            // 537 MB output stream: nontemporal so g/W_out stay hot in L2
            __builtin_nontemporal_store(acc[j][v] + bias, out + row * VOCAB + col);
        }
    }
}

extern "C" void kernel_launch(void* const* d_in, const int* in_sizes, int n_in,
                              void* d_out, int out_size, void* d_ws, size_t ws_size,
                              hipStream_t stream) {
    const float* enc   = (const float*)d_in[0];
    const float* pred  = (const float*)d_in[1];
    const float* Wenc  = (const float*)d_in[2];
    const float* Wpred = (const float*)d_in[3];
    const float* Wout  = (const float*)d_in[4];
    const float* bout  = (const float*)d_in[5];

    char* ws = (char*)d_ws;
    size_t off = 0;
    _Float16* enc_h   = (_Float16*)(ws + off); off += (size_t)B_ * T_ * KIN * 2;
    _Float16* pred_h  = (_Float16*)(ws + off); off += (size_t)B_ * U_ * KIN * 2;
    _Float16* Wenc_h  = (_Float16*)(ws + off); off += (size_t)D_JOINT * KIN * 2;
    _Float16* Wpred_h = (_Float16*)(ws + off); off += (size_t)D_JOINT * KIN * 2;
    _Float16* Wout_h  = (_Float16*)(ws + off); off += (size_t)VOCAB * D_JOINT * 2;
    float*    fbuf    = (float*)(ws + off);    off += (size_t)B_ * T_ * D_JOINT * 4;
    float*    gbuf    = (float*)(ws + off);    off += (size_t)B_ * U_ * D_JOINT * 4;
    // total ~11.8 MB of d_ws

    cvt_f32_to_f16<<<(B_*T_*KIN + 255) / 256, 256, 0, stream>>>(enc, enc_h, B_*T_*KIN);
    cvt_f32_to_f16<<<(B_*U_*KIN + 255) / 256, 256, 0, stream>>>(pred, pred_h, B_*U_*KIN);
    cvt_f32_to_f16<<<(D_JOINT*KIN + 255) / 256, 256, 0, stream>>>(Wenc, Wenc_h, D_JOINT*KIN);
    cvt_f32_to_f16<<<(D_JOINT*KIN + 255) / 256, 256, 0, stream>>>(Wpred, Wpred_h, D_JOINT*KIN);
    cvt_f32_to_f16<<<(VOCAB*D_JOINT + 255) / 256, 256, 0, stream>>>(Wout, Wout_h, VOCAB*D_JOINT);

    const size_t gemm_lds = (size_t)(64 * GP + 128 * GP) * 2;        // 27648 B
    gemm_fg<<<dim3((B_*T_) / 64, D_JOINT / 128), 256, gemm_lds, stream>>>(enc_h,  Wenc_h,  fbuf);
    gemm_fg<<<dim3((B_*U_) / 64, D_JOINT / 128), 256, gemm_lds, stream>>>(pred_h, Wpred_h, gbuf);

    const size_t main_lds = 2560 + (size_t)(2 * 64 * GP + 2 * 256 * GP) * 2; // 94720 B
    joint_gemm<<<dim3(B_ * T_, VOCAB / 256), 256, main_lds, stream>>>(
        fbuf, gbuf, Wout_h, bout, (float*)d_out);
}
